// PewSequenceModel_498216206992
// MI455X (gfx1250) — compile-verified, matrix operand order checked by reading, not run on hardware
//
#include <hip/hip_runtime.h>

typedef __attribute__((ext_vector_type(16))) _Float16 v16h;
typedef __attribute__((ext_vector_type(8)))  float    v8f;

#define BATCH  4096
#define SEQ    2048
#define NELEM  (BATCH * SEQ)      // 8,388,608 (b,t) elements
#define NTILES (NELEM / 16)       // 524,288 WMMA tiles
#define PT     2                  // batch chains per scan thread (ILP)
#define SCAN_THREADS (BATCH / PT) // 2048

// ---------------- fast activations (v_exp_f32 + v_rcp_f32) ----------------
__device__ __forceinline__ float sigm(float x) {
  float e = __expf(-x);
  return __builtin_amdgcn_rcpf(1.0f + e);
}
__device__ __forceinline__ float tanhx(float x) {
  float e = __expf(-2.0f * x);
  return (1.0f - e) * __builtin_amdgcn_rcpf(1.0f + e);
}

// One LSTM cell update (HS==1): gates already contain all non-h terms.
__device__ __forceinline__ void cell_step(float g0, float g1, float g2, float g3,
                                          const float u[4], float& h, float& c) {
  float a0 = fmaf(h, u[0], g0);
  float a1 = fmaf(h, u[1], g1);
  float a2 = fmaf(h, u[2], g2);
  float a3 = fmaf(h, u[3], g3);
  float ig = sigm(a0);
  float fg = sigm(a1);
  float gg = tanhx(a2);
  float og = sigm(a3);
  c = fmaf(fg, c, ig * gg);
  h = og * tanhx(c);
}

// ---------------------------------------------------------------------------
// Kernel 1: WMMA pre-projection of all time-independent gate terms.
//   gpre[e*8 + 0..3] = parking*W1 + weather . V1[:,j] + b1[j]
//   gpre[e*8 + 4..7] = weather . V2[:,j-4]           + b2[j-4]
// One v_wmma_f32_16x16x32_f16 per 16 consecutive elements:
//   A (16x32 f16): row m = [w0,w1,w2,w3,pw, 0...]
//   B (32x16 f16): rows 0..3 = [V1 | V2], row 4 = [W1 | 0], rest 0
//   C (16x16 f32): bias broadcast per column
// ---------------------------------------------------------------------------
__global__ void precompute_gates(const float* __restrict__ x,
                                 const float* __restrict__ W1,
                                 const float* __restrict__ V1,
                                 const float* __restrict__ b1,
                                 const float* __restrict__ V2,
                                 const float* __restrict__ b2,
                                 float* __restrict__ gpre) {
  const int lane   = threadIdx.x & 31;
  const int wave   = (blockIdx.x * blockDim.x + threadIdx.x) >> 5;
  const int nwaves = (gridDim.x * blockDim.x) >> 5;
  const int n      = lane & 15;   // output column N for this lane

  // B fragment: constant across tiles.
  // 16-bit B 32x16 layout: VGPR j, lanes 0-15 hold K=2j,2j+1 => half h = B[K=h][n]
  // lanes 16-31 hold K=8..15,24..31 (all zero here).
  v16h bf = {};
  if (lane < 16 && n < 8) {
#pragma unroll
    for (int k = 0; k < 4; ++k) {
      float v = (n < 4) ? V1[k * 4 + n] : V2[k * 4 + (n - 4)];
      bf[k] = (_Float16)v;
    }
    bf[4] = (n < 4) ? (_Float16)W1[n] : (_Float16)0.0f;
  }
  float bias = 0.0f;
  if (n < 4)      bias = b1[n];
  else if (n < 8) bias = b2[n - 4];

  for (long tile = wave; tile < NTILES; tile += nwaves) {
    const long base = tile * 16;

    // A fragment: 16-bit A 16x32 layout: lanes 0-15 row M=lane, halves 0..7 = K0..7.
    v16h af = {};
    if (lane < 16) {
      const float* xp = x + (base + lane) * 5;
      af[0] = (_Float16)xp[0];
      af[1] = (_Float16)xp[1];
      af[2] = (_Float16)xp[2];
      af[3] = (_Float16)xp[3];
      af[4] = (_Float16)xp[4];
    }

    v8f c;
#pragma unroll
    for (int r = 0; r < 8; ++r) c[r] = bias;

    c = __builtin_amdgcn_wmma_f32_16x16x32_f16(
        /*neg_a=*/false, af, /*neg_b=*/false, bf,
        /*c_mod=*/(short)0, c, /*reuse_a=*/false, /*reuse_b=*/false);

    // D layout: VGPR r -> M=r (lanes 0-15) / M=8+r (lanes 16-31); N = lane&15.
    const int mbase = (lane < 16) ? 0 : 8;
    if (n < 8) {
#pragma unroll
      for (int r = 0; r < 8; ++r) {
        gpre[(base + mbase + r) * 8 + n] = c[r];
      }
    }
  }
}

// ---------------------------------------------------------------------------
// Kernel 2: serial two-layer LSTM scan; PT independent batch chains per thread.
// gpre stream is contiguous per thread (32B/step); outputs buffered to float4.
// ---------------------------------------------------------------------------
__global__ void lstm_scan_gpre(const float* __restrict__ gpre,
                               const float* __restrict__ U1,
                               const float* __restrict__ W2,
                               const float* __restrict__ U2,
                               const float* __restrict__ fc_w,
                               const float* __restrict__ fc_b,
                               float* __restrict__ out) {
  const int tid = blockIdx.x * blockDim.x + threadIdx.x;  // 0 .. SCAN_THREADS-1
  float u1[4], w2[4], u2[4];
#pragma unroll
  for (int j = 0; j < 4; ++j) { u1[j] = U1[j]; w2[j] = W2[j]; u2[j] = U2[j]; }
  const float fw = fc_w[0], fb = fc_b[0];

  float h1[PT], c1[PT], h2[PT], c2[PT];
  long eb[PT];
#pragma unroll
  for (int i = 0; i < PT; ++i) {
    h1[i] = 0.0f; c1[i] = 0.0f; h2[i] = 0.0f; c2[i] = 0.0f;
    eb[i] = ((long)tid + (long)i * SCAN_THREADS) * SEQ;
  }

  for (int t0 = 0; t0 < SEQ; t0 += 4) {
#pragma unroll
    for (int i = 0; i < PT; ++i) {
      const float4* gp = (const float4*)(gpre + (eb[i] + t0) * 8);
      if (t0 + 8 < SEQ)  // gfx1250: global_prefetch_b8 of upcoming gate lines
        __builtin_prefetch(gpre + (eb[i] + t0 + 8) * 8, 0, 0);
      float4 ob;
#pragma unroll
      for (int s = 0; s < 4; ++s) {
        float4 ga = gp[2 * s];
        float4 gb = gp[2 * s + 1];
        cell_step(ga.x, ga.y, ga.z, ga.w, u1, h1[i], c1[i]);
        float q0 = fmaf(h1[i], w2[0], gb.x);
        float q1 = fmaf(h1[i], w2[1], gb.y);
        float q2 = fmaf(h1[i], w2[2], gb.z);
        float q3 = fmaf(h1[i], w2[3], gb.w);
        cell_step(q0, q1, q2, q3, u2, h2[i], c2[i]);
        float o = fmaf(h2[i], fw, fb);
        if (s == 0) ob.x = o; else if (s == 1) ob.y = o;
        else if (s == 2) ob.z = o; else ob.w = o;
      }
      *(float4*)(out + eb[i] + t0) = ob;
    }
  }
}

// ---------------------------------------------------------------------------
// Fallback: fully fused scan (used only if workspace can't hold gpre, 256 MB).
// ---------------------------------------------------------------------------
__global__ void lstm_fused(const float* __restrict__ x,
                           const float* __restrict__ W1, const float* __restrict__ U1,
                           const float* __restrict__ V1, const float* __restrict__ b1,
                           const float* __restrict__ W2, const float* __restrict__ U2,
                           const float* __restrict__ V2, const float* __restrict__ b2,
                           const float* __restrict__ fc_w, const float* __restrict__ fc_b,
                           float* __restrict__ out) {
  const int tid = blockIdx.x * blockDim.x + threadIdx.x;
  float w1[4], u1[4], v1[16], bb1[4], w2[4], u2[4], v2[16], bb2[4];
#pragma unroll
  for (int j = 0; j < 4; ++j) {
    w1[j] = W1[j]; u1[j] = U1[j]; bb1[j] = b1[j];
    w2[j] = W2[j]; u2[j] = U2[j]; bb2[j] = b2[j];
  }
#pragma unroll
  for (int j = 0; j < 16; ++j) { v1[j] = V1[j]; v2[j] = V2[j]; }
  const float fw = fc_w[0], fb = fc_b[0];

  float h1[PT], c1[PT], h2[PT], c2[PT];
  long eb[PT];
#pragma unroll
  for (int i = 0; i < PT; ++i) {
    h1[i] = 0.0f; c1[i] = 0.0f; h2[i] = 0.0f; c2[i] = 0.0f;
    eb[i] = ((long)tid + (long)i * SCAN_THREADS) * SEQ;
  }

  for (int t0 = 0; t0 < SEQ; t0 += 4) {
#pragma unroll
    for (int i = 0; i < PT; ++i) {
      const float* xp = x + (eb[i] + t0) * 5;
      if (t0 + 8 < SEQ)
        __builtin_prefetch(x + (eb[i] + t0 + 8) * 5, 0, 0);
      float4 ob;
#pragma unroll
      for (int s = 0; s < 4; ++s) {
        float wv0 = xp[s * 5 + 0], wv1 = xp[s * 5 + 1];
        float wv2 = xp[s * 5 + 2], wv3 = xp[s * 5 + 3];
        float pw  = xp[s * 5 + 4];
        float g1v[4], g2v[4];
#pragma unroll
        for (int j = 0; j < 4; ++j) {
          float a = fmaf(pw, w1[j], bb1[j]);
          a = fmaf(wv0, v1[j], a);  a = fmaf(wv1, v1[4 + j], a);
          a = fmaf(wv2, v1[8 + j], a); a = fmaf(wv3, v1[12 + j], a);
          g1v[j] = a;
          float b = bb2[j];
          b = fmaf(wv0, v2[j], b);  b = fmaf(wv1, v2[4 + j], b);
          b = fmaf(wv2, v2[8 + j], b); b = fmaf(wv3, v2[12 + j], b);
          g2v[j] = b;
        }
        cell_step(g1v[0], g1v[1], g1v[2], g1v[3], u1, h1[i], c1[i]);
        float q0 = fmaf(h1[i], w2[0], g2v[0]);
        float q1 = fmaf(h1[i], w2[1], g2v[1]);
        float q2 = fmaf(h1[i], w2[2], g2v[2]);
        float q3 = fmaf(h1[i], w2[3], g2v[3]);
        cell_step(q0, q1, q2, q3, u2, h2[i], c2[i]);
        float o = fmaf(h2[i], fw, fb);
        if (s == 0) ob.x = o; else if (s == 1) ob.y = o;
        else if (s == 2) ob.z = o; else ob.w = o;
      }
      *(float4*)(out + eb[i] + t0) = ob;
    }
  }
}

extern "C" void kernel_launch(void* const* d_in, const int* in_sizes, int n_in,
                              void* d_out, int out_size, void* d_ws, size_t ws_size,
                              hipStream_t stream) {
  const float* x    = (const float*)d_in[0];
  const float* W1   = (const float*)d_in[1];
  const float* U1   = (const float*)d_in[2];
  const float* V1   = (const float*)d_in[3];
  const float* b1   = (const float*)d_in[4];
  const float* W2   = (const float*)d_in[5];
  const float* U2   = (const float*)d_in[6];
  const float* V2   = (const float*)d_in[7];
  const float* b2   = (const float*)d_in[8];
  const float* fc_w = (const float*)d_in[9];
  const float* fc_b = (const float*)d_in[10];
  float* out = (float*)d_out;

  const size_t need = (size_t)NELEM * 8 * sizeof(float);  // 256 MB gate buffer
  if (ws_size >= need) {
    float* gpre = (float*)d_ws;
    precompute_gates<<<1024, 256, 0, stream>>>(x, W1, V1, b1, V2, b2, gpre);
    lstm_scan_gpre<<<SCAN_THREADS / 64, 64, 0, stream>>>(gpre, U1, W2, U2, fc_w, fc_b, out);
  } else {
    lstm_fused<<<SCAN_THREADS / 64, 64, 0, stream>>>(x, W1, U1, V1, b1, W2, U2, V2, b2,
                                                     fc_w, fc_b, out);
  }
}
